// ParCA_23897198035556
// MI455X (gfx1250) — matrix operand order repaired and checked by reading, not compiled, not run on hardware
//
#include <hip/hip_runtime.h>

// ---------------- WMMA tile constants ----------------
#define BM 128
#define BN 128
#define BK 32
#define KPAD 40   // row stride in ushorts: 80B = 16B-aligned for ds_load_b128, conflict-friendly

typedef __attribute__((ext_vector_type(16))) __bf16 v16bf;
typedef __attribute__((ext_vector_type(8)))  float  v8f;

__device__ __forceinline__ unsigned short f2bf(float f) {
  union { __bf16 h; unsigned short u; } a; a.h = (__bf16)f; return a.u;
}
__device__ __forceinline__ unsigned pk2bf(float x, float y) {
  union { __bf16 h[2]; unsigned u; } a; a.h[0] = (__bf16)x; a.h[1] = (__bf16)y; return a.u;
}
__device__ __forceinline__ float bf2f(unsigned short h) {
  union { unsigned u; float f; } a; a.u = ((unsigned)h) << 16; return a.f;
}
__device__ __forceinline__ unsigned lds_off(const void* p) {
  return (unsigned)(unsigned long long)p;   // low 32 bits of generic addr = LDS offset
}
// async copy of 16 bytes global -> LDS (tracked by ASYNCcnt)
__device__ __forceinline__ void async_b128(unsigned loff, const void* g) {
  asm volatile("global_load_async_to_lds_b128 %0, %1, off" :: "v"(loff), "v"(g) : "memory");
}
__device__ __forceinline__ void wait_async0() {
  asm volatile("s_wait_asynccnt 0x0" ::: "memory");
}

// A fragment 16x32 bf16: per lane-half two contiguous 16B runs -> 2x ds_load_b128
__device__ __forceinline__ v16bf frag_a(const unsigned short* row, int hl) {
  union { uint4 q[2]; v16bf v; } f;
  f.q[0] = *reinterpret_cast<const uint4*>(row + hl * 8);
  f.q[1] = *reinterpret_cast<const uint4*>(row + 16 + hl * 8);
  return f.v;
}
// B fragment 32x16 bf16: per lane-half two contiguous 16B runs -> 2x ds_load_b128
__device__ __forceinline__ v16bf frag_b(const unsigned short* row, int hl) {
  union { uint4 q[2]; v16bf v; } f;
  const unsigned short* p = row + hl * 8;
  f.q[0] = *reinterpret_cast<const uint4*>(p);
  f.q[1] = *reinterpret_cast<const uint4*>(p + 16);
  return f.v;
}

// ======== Universal NT GEMM: C[M,N] = A[M,K] x Bt[N,K]^T, bf16 in, all-async staging ========
// Waves in 4x2 grid: each computes 32x64 (2 A-frags x 4 B-frags = 8 WMMA/K-step).
// MODE 0: bf16 out, += bias[n].   MODE 1: f32 out, *scale + (mask[b][n] | addc).
template<int MODE>
__global__ __launch_bounds__(256)
void gemm_nt(const unsigned short* __restrict__ A, int lda, long long sA,
             const unsigned short* __restrict__ Bt, int ldb, long long sB,
             void* __restrict__ OutP, int ldo, long long sO,
             const float* __restrict__ bias,
             const float* __restrict__ mask, long long sM,
             float scale, float addc, int K) {
  __shared__ __align__(16) unsigned short As[2][BM * KPAD];
  __shared__ __align__(16) unsigned short Bs[2][BN * KPAD];
  const int b = blockIdx.z;
  const unsigned short* Ab = A + (size_t)b * sA;
  const unsigned short* Bb = Bt + (size_t)b * sB;
  const int tid = threadIdx.x;
  const int lane = tid & 31, wv = tid >> 5;
  const int hl = lane >> 4, l16 = lane & 15;
  const int mbase = (wv & 3) * 32, nbase = (wv >> 2) * 64;
  const int m0 = blockIdx.y * BM, n0 = blockIdx.x * BN;

  v8f acc[2][4];
#pragma unroll
  for (int i = 0; i < 2; ++i)
#pragma unroll
    for (int t = 0; t < 4; ++t)
#pragma unroll
      for (int j = 0; j < 8; ++j) acc[i][t][j] = 0.0f;

  auto stage = [&](int buf, int k0) {
#pragma unroll
    for (int e = 0; e < 2; ++e) {
      const int idx8 = tid * 2 + e;                  // 512 chunks each for A and B
      const int row = idx8 >> 2, col8 = (idx8 & 3) << 3;
      async_b128(lds_off(&As[buf][row * KPAD + col8]),
                 &Ab[(size_t)(m0 + row) * lda + k0 + col8]);
      async_b128(lds_off(&Bs[buf][row * KPAD + col8]),
                 &Bb[(size_t)(n0 + row) * ldb + k0 + col8]);
    }
  };

  auto compute = [&](int buf) {
    v16bf af[2], bfv[4];
#pragma unroll
    for (int mi = 0; mi < 2; ++mi)
      af[mi] = frag_a(&As[buf][(mbase + mi * 16 + l16) * KPAD], hl);
#pragma unroll
    for (int nt = 0; nt < 4; ++nt)
      bfv[nt] = frag_b(&Bs[buf][(nbase + nt * 16 + l16) * KPAD], hl);
#pragma unroll
    for (int mi = 0; mi < 2; ++mi)
#pragma unroll
      for (int nt = 0; nt < 4; ++nt)
        acc[mi][nt] = __builtin_amdgcn_wmma_f32_16x16x32_bf16(
            false, af[mi], false, bfv[nt], (short)0, acc[mi][nt], false, false);
  };

  stage(0, 0);
  wait_async0();
  __syncthreads();

  const int nT = K / BK;                 // always even here (32 or 64)
  for (int t = 0; t < nT; t += 2) {
    if (t + 1 < nT) stage(1, (t + 1) * BK);
    compute(0);
    wait_async0();
    __syncthreads();
    if (t + 2 < nT) stage(0, (t + 2) * BK);
    compute(1);
    wait_async0();
    __syncthreads();
  }

  if (MODE == 0) {
    unsigned short* O = (unsigned short*)OutP + (size_t)b * sO;
#pragma unroll
    for (int mi = 0; mi < 2; ++mi)
#pragma unroll
      for (int nt = 0; nt < 4; ++nt) {
        const int n = n0 + nbase + nt * 16 + l16;
        const float bv = bias[n];
#pragma unroll
        for (int v = 0; v < 8; ++v) {
          const int m = m0 + mbase + mi * 16 + hl * 8 + v;
          O[(size_t)m * ldo + n] = f2bf(acc[mi][nt][v] + bv);
        }
      }
  } else {
    float* O = (float*)OutP + (size_t)b * sO;
    const float* mb = mask ? mask + (size_t)b * sM : nullptr;
#pragma unroll
    for (int mi = 0; mi < 2; ++mi)
#pragma unroll
      for (int nt = 0; nt < 4; ++nt) {
        const int n = n0 + nbase + nt * 16 + l16;
        const float add = mb ? mb[n] : addc;
#pragma unroll
        for (int v = 0; v < 8; ++v) {
          const int m = m0 + mbase + mi * 16 + hl * 8 + v;
          O[(size_t)m * ldo + n] = acc[mi][nt][v] * scale + add;
        }
      }
  }
}

// ---------------- one-shot f32 -> bf16 conversion (vectorized) ----------------
__global__ __launch_bounds__(256)
void cvt_bf16(const float* __restrict__ x, unsigned short* __restrict__ o) {
  const size_t i = ((size_t)blockIdx.x * 256 + threadIdx.x) * 4;
  const float4 v = *reinterpret_cast<const float4*>(&x[i]);
  unsigned* d = reinterpret_cast<unsigned*>(&o[i]);
  d[0] = pk2bf(v.x, v.y); d[1] = pk2bf(v.z, v.w);
}

// ---------------- one-shot weight transpose: W[K,N] f32 -> WT[N,K] bf16 ----------------
__global__ __launch_bounds__(256)
void transpose_wf32(const float* __restrict__ W, unsigned short* __restrict__ WT,
                    int Kd, int Nd) {
  __shared__ unsigned short t[32][33];
  const int k0 = blockIdx.y * 32, n0 = blockIdx.x * 32;
  const int tid = threadIdx.x;
#pragma unroll
  for (int e = 0; e < 4; ++e) {
    const int idx = tid + e * 256;
    const int k = idx >> 5, n = idx & 31;
    t[n][k] = f2bf(W[(size_t)(k0 + k) * Nd + n0 + n]);   // coalesced over n
  }
  __syncthreads();
#pragma unroll
  for (int e = 0; e < 4; ++e) {
    const int idx = tid + e * 256;
    const int n = idx >> 5, k = idx & 31;
    WT[(size_t)(n0 + n) * Kd + k0 + k] = t[n][k];        // coalesced over k
  }
}

// ---------------- batched bf16 transpose: V[R,C] -> VT[C,R] ----------------
__global__ __launch_bounds__(256)
void transpose_u16(const unsigned short* __restrict__ in, unsigned short* __restrict__ out,
                   int R, int C) {
  const int b = blockIdx.z;
  in  += (size_t)b * R * C;
  out += (size_t)b * R * C;
  __shared__ unsigned short t[32][33];
  const int r0 = blockIdx.y * 32, c0 = blockIdx.x * 32;
  const int tid = threadIdx.x;
#pragma unroll
  for (int e = 0; e < 4; ++e) {
    const int idx = tid + e * 256;
    const int r = idx >> 5, c = idx & 31;
    t[c][r] = in[(size_t)(r0 + r) * C + c0 + c];
  }
  __syncthreads();
#pragma unroll
  for (int e = 0; e < 4; ++e) {
    const int idx = tid + e * 256;
    const int c = idx >> 5, r = idx & 31;
    out[(size_t)(c0 + c) * R + r0 + r] = t[c][r];
  }
}

// ---------------- softmax over each row of f32 scores; write bf16 P in place ----------------
__global__ __launch_bounds__(256)
void softmax_rows(float* __restrict__ Sc, int S_) {
  float* row = Sc + (size_t)blockIdx.x * S_;
  unsigned short* prow = reinterpret_cast<unsigned short*>(row);
  __shared__ float red[256];
  const int tid = threadIdx.x;
  float vals[8];
  float mx = -3.4e38f;
#pragma unroll
  for (int e = 0; e < 8; ++e) { vals[e] = row[tid + e * 256]; mx = fmaxf(mx, vals[e]); }
  red[tid] = mx; __syncthreads();
  for (int s = 128; s > 0; s >>= 1) { if (tid < s) red[tid] = fmaxf(red[tid], red[tid + s]); __syncthreads(); }
  mx = red[0]; __syncthreads();
  float sum = 0.0f;
#pragma unroll
  for (int e = 0; e < 8; ++e) { vals[e] = __expf(vals[e] - mx); sum += vals[e]; }
  red[tid] = sum; __syncthreads();
  for (int s = 128; s > 0; s >>= 1) { if (tid < s) red[tid] += red[tid + s]; __syncthreads(); }
  const float inv = 1.0f / red[0];
#pragma unroll
  for (int e = 0; e < 8; ++e) prow[tid + e * 256] = f2bf(vals[e] * inv);
}

// ---------------- residual + LayerNorm -> bf16 out (feeds projections only) ----------------
__global__ __launch_bounds__(256)
void res_layernorm(const float* __restrict__ x, const float* __restrict__ att,
                   const float* __restrict__ g, const float* __restrict__ bt,
                   unsigned short* __restrict__ out, int Dk) {
  const size_t roff = (size_t)blockIdx.x * Dk;
  __shared__ float red[256];
  const int tid = threadIdx.x;
  float v[4]; float s = 0.0f;
#pragma unroll
  for (int e = 0; e < 4; ++e) { const int c = tid + e * 256; v[e] = x[roff + c] + att[roff + c]; s += v[e]; }
  red[tid] = s; __syncthreads();
  for (int st = 128; st > 0; st >>= 1) { if (tid < st) red[tid] += red[tid + st]; __syncthreads(); }
  const float mean = red[0] * (1.0f / 1024.0f); __syncthreads();
  float var = 0.0f;
#pragma unroll
  for (int e = 0; e < 4; ++e) { const float d = v[e] - mean; var += d * d; }
  red[tid] = var; __syncthreads();
  for (int st = 128; st > 0; st >>= 1) { if (tid < st) red[tid] += red[tid + st]; __syncthreads(); }
  const float rinv = rsqrtf(red[0] * (1.0f / 1024.0f) + 1e-5f);
#pragma unroll
  for (int e = 0; e < 4; ++e) {
    const int c = tid + e * 256;
    out[roff + c] = f2bf((v[e] - mean) * rinv * g[c] + bt[c]);
  }
}

// ---------------- column-sum of softmax P (sum over q) ----------------
__global__ __launch_bounds__(256)
void colsum_kernel(const unsigned short* __restrict__ P, float* __restrict__ cs, int S_) {
  const int b = blockIdx.y;
  const int k = blockIdx.x * 256 + threadIdx.x;
  const unsigned short* Pb = P + (size_t)b * S_ * 2 * S_;
  float s = 0.0f;
  for (int q = 0; q < S_; ++q) s += bf2f(Pb[(size_t)q * 2 * S_ + k]);
  cs[b * S_ + k] = s;
}

// ---------------- att[b,d] = sum_k cs[b,k] * V[b,k,d] ----------------
__global__ __launch_bounds__(256)
void attsum_kernel(const float* __restrict__ cs, const unsigned short* __restrict__ V,
                   float* __restrict__ att, int S_, int Dk) {
  const int b = blockIdx.y;
  const int d = blockIdx.x * 256 + threadIdx.x;
  const unsigned short* Vb = V + (size_t)b * S_ * Dk;
  const float* cb = cs + b * S_;
  float s = 0.0f;
  for (int k = 0; k < S_; ++k) s += cb[k] * bf2f(Vb[(size_t)k * Dk + d]);
  att[b * Dk + d] = s;
}

// ---------------- final concat + linear ----------------
__global__ __launch_bounds__(256)
void final_linear(const float* __restrict__ a1, const float* __restrict__ a2,
                  const float* __restrict__ Wl, const float* __restrict__ bl,
                  float* __restrict__ out, int Dk) {
  const int b = blockIdx.y;
  const int n = blockIdx.x * 256 + threadIdx.x;
  const float* x1 = a1 + b * Dk;
  const float* x2 = a2 + b * Dk;
  float s = bl[n];
  for (int d = 0; d < Dk; ++d) s += x1[d] * Wl[(size_t)d * Dk + n];
  for (int d = 0; d < Dk; ++d) s += x2[d] * Wl[(size_t)(Dk + d) * Dk + n];
  out[b * Dk + n] = s;
}

// ---------------- host orchestration ----------------
extern "C" void kernel_launch(void* const* d_in, const int* in_sizes, int n_in,
                              void* d_out, int out_size, void* d_ws, size_t ws_size,
                              hipStream_t stream) {
  (void)in_sizes; (void)n_in; (void)out_size; (void)ws_size;
  const int B_ = 8, S = 2048, D = 1024;
  const int M = B_ * S;  // 16384
  typedef long long ll;

  const float* modalf[3] = {(const float*)d_in[0], (const float*)d_in[2], (const float*)d_in[4]};
  const float* mask2 = (const float*)d_in[3];
  const float* mask3 = (const float*)d_in[5];
  const float* Wt[4][3]; const float* Bi[4][3];
  for (int blk = 0; blk < 4; ++blk)
    for (int p = 0; p < 3; ++p) {
      Wt[blk][p] = (const float*)d_in[6 + blk * 6 + p * 2];
      Bi[blk][p] = (const float*)d_in[6 + blk * 6 + p * 2 + 1];
    }
  const float* lin_w = (const float*)d_in[30];
  const float* lin_b = (const float*)d_in[31];
  const float* n_g[2] = {(const float*)d_in[32], (const float*)d_in[34]};
  const float* n_b[2] = {(const float*)d_in[33], (const float*)d_in[35]};

  char* ws = (char*)d_ws; size_t off = 0;
  auto alloc = [&](size_t bytes) -> char* {
    char* p = ws + off; off += (bytes + 255) & ~(size_t)255; return p;
  };
  unsigned short* mb[3];
  for (int i = 0; i < 3; ++i) mb[i] = (unsigned short*)alloc((size_t)M * D * 2);
  unsigned short* WTbuf = (unsigned short*)alloc((size_t)12 * D * D * 2);
  unsigned short* Qb = (unsigned short*)alloc((size_t)M * D * 2);
  unsigned short* Kb = (unsigned short*)alloc((size_t)M * D * 2);
  unsigned short* Vb = (unsigned short*)alloc((size_t)M * D * 2);
  unsigned short* VT = (unsigned short*)alloc((size_t)M * D * 2);
  float* Sc   = (float*)alloc((size_t)B_ * S * S * 4);  // f32 scores; bf16 P written in place
  float* attn = (float*)alloc((size_t)M * D * 4);
  unsigned short* cab = (unsigned short*)alloc((size_t)M * D * 2);
  float* csum = (float*)alloc((size_t)B_ * S * 4);
  float* att1 = (float*)alloc((size_t)B_ * D * 4);
  float* att2 = (float*)alloc((size_t)B_ * D * 4);

  // ---- pre-pass: convert activations, transpose+convert all weights (one-shot) ----
  for (int i = 0; i < 3; ++i)
    cvt_bf16<<<(M * D) / 1024, 256, 0, stream>>>(modalf[i], mb[i]);
  for (int blk = 0; blk < 4; ++blk)
    for (int p = 0; p < 3; ++p)
      transpose_wf32<<<dim3(32, 32), 256, 0, stream>>>(
          Wt[blk][p], WTbuf + (size_t)(blk * 3 + p) * D * D, D, D);
  auto WTs = [&](int blk, int p) { return WTbuf + (size_t)(blk * 3 + p) * D * D; };

  const float inv_scale = 1.0f / 32.0f;  // 1/sqrt(1024)
  const dim3 gProj(D / BN, M / BM, 1);       // (8,128,1)
  const dim3 gScores(S / BN, S / BM, B_);    // (16,16,8)
  const dim3 gPV(D / BN, S / BM, B_);        // (8,16,8)

  for (int br = 0; br < 2; ++br) {
    const unsigned short* ctxb = (br == 0) ? mb[1] : mb[2];
    const float* mask = (br == 0) ? mask2 : mask3;
    const int caI = br, aI = 2 + br;
    float* attOut = (br == 0) ? att1 : att2;

    // Cross-attention projections (bf16 NT, async)
    gemm_nt<0><<<gProj, 256, 0, stream>>>(mb[0], D, 0, WTs(caI, 0), D, 0, Qb, D, 0,
                                          Bi[caI][0], nullptr, 0, 0.f, 0.f, D);
    gemm_nt<0><<<gProj, 256, 0, stream>>>(ctxb, D, 0, WTs(caI, 1), D, 0, Kb, D, 0,
                                          Bi[caI][1], nullptr, 0, 0.f, 0.f, D);
    gemm_nt<0><<<gProj, 256, 0, stream>>>(ctxb, D, 0, WTs(caI, 2), D, 0, Vb, D, 0,
                                          Bi[caI][2], nullptr, 0, 0.f, 0.f, D);
    // Scores, softmax, V-transpose, P@V
    gemm_nt<1><<<gScores, 256, 0, stream>>>(Qb, D, (ll)S * D, Kb, D, (ll)S * D,
                                            Sc, S, (ll)S * S, nullptr,
                                            mask, S, inv_scale, 0.0f, D);
    softmax_rows<<<M, 256, 0, stream>>>(Sc, S);
    transpose_u16<<<dim3(D / 32, S / 32, B_), 256, 0, stream>>>(Vb, VT, S, D);
    gemm_nt<1><<<gPV, 256, 0, stream>>>((const unsigned short*)Sc, 2 * S, (ll)S * 2 * S,
                                        VT, S, (ll)D * S, attn, D, (ll)S * D,
                                        nullptr, nullptr, 0, 1.0f, 0.0f, S);
    // Residual + LayerNorm -> bf16
    res_layernorm<<<M, 256, 0, stream>>>(modalf[0], attn, n_g[br], n_b[br], cab, D);
    // Self-attention head (summed over q)
    gemm_nt<0><<<gProj, 256, 0, stream>>>(cab, D, 0, WTs(aI, 0), D, 0, Qb, D, 0,
                                          Bi[aI][0], nullptr, 0, 0.f, 0.f, D);
    gemm_nt<0><<<gProj, 256, 0, stream>>>(cab, D, 0, WTs(aI, 1), D, 0, Kb, D, 0,
                                          Bi[aI][1], nullptr, 0, 0.f, 0.f, D);
    gemm_nt<0><<<gProj, 256, 0, stream>>>(cab, D, 0, WTs(aI, 2), D, 0, Vb, D, 0,
                                          Bi[aI][2], nullptr, 0, 0.f, 0.f, D);
    gemm_nt<1><<<gScores, 256, 0, stream>>>(Qb, D, (ll)S * D, Kb, D, (ll)S * D,
                                            Sc, S, (ll)S * S, nullptr,
                                            nullptr, 0, inv_scale, 1.0f, D);
    softmax_rows<<<M, 256, 0, stream>>>(Sc, S);
    colsum_kernel<<<dim3(S / 256, B_), 256, 0, stream>>>((const unsigned short*)Sc, csum, S);
    attsum_kernel<<<dim3(D / 256, B_), 256, 0, stream>>>(csum, Vb, attOut, S, D);
  }
  final_linear<<<dim3(D / 256, B_), 256, 0, stream>>>(att1, att2, lin_w, lin_b, (float*)d_out, D);
}
